// WanSLAProcessor_82334523064717
// MI455X (gfx1250) — matrix operand order repaired and checked by reading, not compiled, not run on hardware
//
#include <hip/hip_runtime.h>

// ---------------- problem constants ----------------
#define SEQ   2048
#define NH    12
#define HD    128
#define DIMSZ 1536
#define BLKQ  128
#define BLKK  64
#define NQB   16          // SEQ/BLKQ
#define NKB   32          // SEQ/BLKK
#define KSEL  7           // ceil(0.2*32)
#define SCALE 0.08838834764831845f   // 1/sqrt(128)
#define EPSV  1e-6f
#define NEGV  -1e30f

typedef __bf16 bf16_t;
typedef __attribute__((ext_vector_type(16))) __bf16 v16bf;
typedef __attribute__((ext_vector_type(8)))  __bf16 v8bf;
typedef __attribute__((ext_vector_type(8)))  float  v8f;

// ---------------- WMMA helper (16x16x32 bf16, f32 accum) ----------------
static __device__ __forceinline__ v8f wmma_bf16(v16bf a, v16bf b, v8f c) {
  return __builtin_amdgcn_wmma_f32_16x16x32_bf16(false, a, false, b, (short)0, c,
                                                 false, false);
}

static __device__ __forceinline__ v16bf make_frag(v8bf lo, v8bf hi) {
  union { v16bf v; struct { v8bf lo, hi; } p; } u;
  u.p.lo = lo;
  u.p.hi = hi;
  return u.v;
}

// A fragment: 16x32 bf16 tile, row-major, stride lda.
// lane 0-15: M=lane, elems [0..7]=K(kb..kb+7), [8..15]=K(16+kb..23+kb), kb=0
// lane 16-31: same rows, kb=8.  -> two contiguous 16B chunks per lane.
static __device__ __forceinline__ v16bf frag_a(const bf16_t* tile, int lda, int lane) {
  int m  = lane & 15;
  int kb = (lane & 16) ? 8 : 0;
  const bf16_t* p = tile + (size_t)m * lda + kb;
  v8bf lo = *(const v8bf*)p;
  v8bf hi = *(const v8bf*)(p + 16);
  return make_frag(lo, hi);
}

// B fragment from K-contiguous source T where B[k][n] = T[n][k] (stride ldt).
// lane n holds K=kb..kb+7 and 16+kb..23+kb (kb=0 / 8) -> two contiguous 16B chunks.
static __device__ __forceinline__ v16bf frag_bt(const bf16_t* tile, int ldt, int lane) {
  int n  = lane & 15;
  int kb = (lane & 16) ? 8 : 0;
  const bf16_t* p = tile + (size_t)n * ldt + kb;
  v8bf lo = *(const v8bf*)p;
  v8bf hi = *(const v8bf*)(p + 16);
  return make_frag(lo, hi);
}

// ---------------- async global -> LDS (CDNA5, ASYNCcnt) ----------------
static __device__ __forceinline__ void async_b128(unsigned lds_off, unsigned goff,
                                                  const void* sbase) {
  asm volatile("global_load_async_to_lds_b128 %0, %1, %2 offset:0"
               :: "v"(lds_off), "v"(goff), "s"(sbase)
               : "memory");
}
#define WAIT_ASYNC_ALL()  asm volatile("s_wait_asynccnt 0x0" ::: "memory")
#define WAIT_ASYNC_3()    asm volatile("s_wait_asynccnt 0x3" ::: "memory")

static __device__ __forceinline__ unsigned lds_addr_of(const void* p) {
  return (unsigned)(uintptr_t)p;  // low 32 bits of generic ptr = LDS offset
}

// ---------------- elementwise f32 -> bf16 ----------------
__global__ void cvt_bf16_kernel(const float* __restrict__ src,
                                bf16_t* __restrict__ dst, int n) {
  int i = blockIdx.x * blockDim.x + threadIdx.x;
  int stride = gridDim.x * blockDim.x;
  for (; i < n; i += stride) dst[i] = (bf16_t)src[i];
}

// ------- f32 [K][N] -> bf16 transposed [N][K] (tiled transpose) ----------
__global__ __launch_bounds__(256) void cvt_transpose_kernel(
    const float* __restrict__ src, bf16_t* __restrict__ dst, int K, int N) {
  __shared__ float tile[32][33];
  int kb = blockIdx.x * 32, nb = blockIdx.y * 32;
  int tx = threadIdx.x & 31, ty = threadIdx.x >> 5;  // 32 x 8
  for (int i = ty; i < 32; i += 8)
    tile[i][tx] = src[(size_t)(kb + i) * N + nb + tx];
  __syncthreads();
  for (int i = ty; i < 32; i += 8)
    dst[(size_t)(nb + i) * K + kb + tx] = (bf16_t)tile[tx][i];
}

// ---------------- GEMM: C[M][N] = A[M][K] * Bt[N][K]^T + bias[N] ----------
// 256 threads (8 waves). Block tile 128(M) x 64(N); wave stripe 16(M) x 64(N).
// A and B tiles double-buffered in LDS via async global->LDS b128.
__global__ __launch_bounds__(256) void gemm_bias_kernel(
    const bf16_t* __restrict__ A, const bf16_t* __restrict__ Bt,
    const float* __restrict__ bias, float* __restrict__ C,
    int M, int N, int K) {
  __shared__ __align__(16) bf16_t Als[2][128][32];
  __shared__ __align__(16) bf16_t Bls[2][64][32];
  int t = threadIdx.x;
  int lane = t & 31;
  int wave = t >> 5;
  int mblk = blockIdx.y * 128;
  int n0 = blockIdx.x * 64;
  int m0 = wave * 16;
  v8f acc[4] = {};

  // per-thread staging assignment
  int ra = t >> 1, ha = t & 1;       // A: 128 rows x (2 x 32B)
  int rb = t >> 2, cb = t & 3;       // B: 64 rows x (4 x 16B)
  unsigned ldsA0 = lds_addr_of(&Als[0][0][0]);
  unsigned ldsB0 = lds_addr_of(&Bls[0][0][0]);

  auto stage = [&](int k0, int buf) {
    unsigned goA = (unsigned)(((size_t)(mblk + ra) * K + k0) * 2) + ha * 32u;
    unsigned laA = ldsA0 + (unsigned)buf * sizeof(bf16_t) * 128 * 32 + ra * 64u + ha * 32u;
    async_b128(laA, goA, A);
    async_b128(laA + 16u, goA + 16u, A);
    unsigned goB = (unsigned)(((size_t)(n0 + rb) * K + k0) * 2) + cb * 16u;
    unsigned laB = ldsB0 + (unsigned)buf * sizeof(bf16_t) * 64 * 32 + rb * 64u + cb * 16u;
    async_b128(laB, goB, Bt);
  };

  stage(0, 0);
  int nkt = K / 32;
  for (int kt = 0; kt < nkt; ++kt) {
    int cur = kt & 1;
    if (kt + 1 < nkt) {
      stage((kt + 1) * 32, cur ^ 1);
      WAIT_ASYNC_3();
    } else {
      WAIT_ASYNC_ALL();
    }
    __syncthreads();
    v16bf a = frag_a(&Als[cur][m0][0], 32, lane);
#pragma unroll
    for (int nt = 0; nt < 4; ++nt) {
      v16bf b = frag_bt(&Bls[cur][nt * 16][0], 32, lane);
      acc[nt] = wmma_bf16(a, b, acc[nt]);
    }
    __syncthreads();
  }

  int mr = (lane & 16) ? 8 : 0;
  int n = lane & 15;
#pragma unroll
  for (int nt = 0; nt < 4; ++nt)
#pragma unroll
    for (int r = 0; r < 8; ++r) {
      int gm = mblk + m0 + mr + r;
      int gn = n0 + nt * 16 + n;
      C[(size_t)gm * N + gn] = acc[nt][r] + bias[gn];
    }
}

// ------------- RMSNorm (over DIM) + RoPE, reorder to [H][S][D], f32+bf16 ------
__global__ __launch_bounds__(256) void qk_post_kernel(
    const float* __restrict__ proj, const float* __restrict__ g,
    const float* __restrict__ fcos, const float* __restrict__ fsin,
    float* __restrict__ outf, bf16_t* __restrict__ outb) {
  int s = blockIdx.x;
  int t = threadIdx.x;
  __shared__ float red[256];
  __shared__ float rowbuf[DIMSZ];
  float ss = 0.f;
  for (int i = t; i < DIMSZ; i += 256) {
    float x = proj[(size_t)s * DIMSZ + i];
    rowbuf[i] = x;
    ss += x * x;
  }
  red[t] = ss;
  __syncthreads();
  for (int o = 128; o > 0; o >>= 1) {
    if (t < o) red[t] += red[t + o];
    __syncthreads();
  }
  float inv = rsqrtf(red[0] / (float)DIMSZ + EPSV);
  for (int pp = t; pp < DIMSZ / 2; pp += 256) {
    int hh = pp >> 6;
    int p  = pp & 63;
    int i0 = hh * HD + 2 * p;
    float x1 = rowbuf[i0] * inv * g[i0];
    float x2 = rowbuf[i0 + 1] * inv * g[i0 + 1];
    float c  = fcos[(size_t)s * HD + 2 * p];
    float sn = fsin[(size_t)s * HD + 2 * p + 1];
    float e  = x1 * c - x2 * sn;
    float o2 = x1 * sn + x2 * c;
    size_t base = ((size_t)hh * SEQ + s) * HD;
    outf[base + 2 * p]     = e;
    outf[base + 2 * p + 1] = o2;
    outb[base + 2 * p]     = (bf16_t)e;
    outb[base + 2 * p + 1] = (bf16_t)o2;
  }
}

// ------------- V: reorder [S][DIM] -> transposed [H][D][S] bf16 ----------------
__global__ __launch_bounds__(256) void v_post_kernel(const float* __restrict__ proj,
                                                     bf16_t* __restrict__ vt) {
  int s = blockIdx.x;
  for (int i = threadIdx.x; i < DIMSZ; i += 256) {
    int hh = i >> 7, d = i & 127;
    vt[((size_t)hh * HD + d) * SEQ + s] = (bf16_t)proj[(size_t)s * DIMSZ + i];
  }
}

// ------------- softmax over D=128 (lq / lk), bf16 out ----------------
__global__ __launch_bounds__(128) void rowsm_kernel(const float* __restrict__ src,
                                                    bf16_t* __restrict__ dst) {
  int row = blockIdx.x;  // h*SEQ + s
  int t = threadIdx.x;
  __shared__ float red[128];
  float x = src[(size_t)row * HD + t];
  red[t] = x;
  __syncthreads();
  for (int o = 64; o > 0; o >>= 1) {
    if (t < o) red[t] = fmaxf(red[t], red[t + o]);
    __syncthreads();
  }
  float mx = red[0];
  __syncthreads();
  float e = __expf(x - mx);
  red[t] = e;
  __syncthreads();
  for (int o = 64; o > 0; o >>= 1) {
    if (t < o) red[t] += red[t + o];
    __syncthreads();
  }
  dst[(size_t)row * HD + t] = (bf16_t)(e / red[0]);
}

// ------------- block-mean pooling: [H][S][D] -> [H][nblk][D] ----------------
__global__ __launch_bounds__(128) void pool_kernel(const float* __restrict__ src,
                                                   float* __restrict__ dst,
                                                   int rows, int nblk) {
  int b  = blockIdx.x;
  int hh = b / nblk, ib = b % nblk;
  int d  = threadIdx.x;
  const float* p = src + ((size_t)hh * SEQ + (size_t)ib * rows) * HD + d;
  float sum = 0.f;
  for (int r = 0; r < rows; ++r) sum += p[(size_t)r * HD];
  dst[(size_t)b * HD + d] = sum / (float)rows;
}

// ------------- top-7 block selection per (h, qblock) -> uint32 bitmask --------
__global__ __launch_bounds__(32) void topk_kernel(const float* __restrict__ qb,
                                                  const float* __restrict__ kb,
                                                  unsigned* __restrict__ maskbits) {
  int hh = blockIdx.x;
  int i  = threadIdx.x;
  if (i >= NQB) return;
  float sc[NKB];
  const float* qp = qb + ((size_t)hh * NQB + i) * HD;
  for (int j = 0; j < NKB; ++j) {
    const float* kp = kb + ((size_t)hh * NKB + j) * HD;
    float s = 0.f;
    for (int d = 0; d < HD; ++d) s += qp[d] * kp[d];
    sc[j] = s;
  }
  unsigned mask = 0;
  for (int t = 0; t < KSEL; ++t) {
    int best = 0;
    float bv = -3.4e38f;
    for (int j = 0; j < NKB; ++j)
      if (!((mask >> j) & 1u) && sc[j] > bv) { bv = sc[j]; best = j; }
    mask |= 1u << best;
  }
  maskbits[hh * NQB + i] = mask;
}

// ------------- sparse-linear attention, flash-style per (h, qblock) ----------
// v is consumed transposed: vt[H][D][S] so B fragments are K-contiguous.
__global__ __launch_bounds__(256) void sla_kernel(
    const bf16_t* __restrict__ qb16, const bf16_t* __restrict__ kb16,
    const bf16_t* __restrict__ vt, const bf16_t* __restrict__ lq16,
    const bf16_t* __restrict__ lk16, const unsigned* __restrict__ maskbits,
    bf16_t* __restrict__ obf) {  // [S][DIM]
  constexpr int STS = 68;   // f32 score tile stride
  constexpr int PTS = 80;   // bf16 prob tile stride (keeps 16B alignment)
  __shared__ float s_tile[BLKQ][STS];
  __shared__ __align__(16) bf16_t p16[BLKQ][PTS];
  __shared__ float s_m[BLKQ], s_ds[BLKQ], s_dl[BLKQ], s_rs[BLKQ];
  int hh = blockIdx.x, qblk = blockIdx.y;
  int lane = threadIdx.x & 31, wave = threadIdx.x >> 5;
  int t = threadIdx.x;
  unsigned mask = maskbits[hh * NQB + qblk];
  int qrow0 = qblk * BLKQ;
  int m0 = wave * 16;
  if (t < BLKQ) { s_m[t] = NEGV; s_ds[t] = 0.f; s_dl[t] = 0.f; s_rs[t] = 1.f; }
  v8f acc_s[8] = {};
  v8f acc_l[8] = {};
  __syncthreads();

  for (int kb = 0; kb < NKB; ++kb) {
    bool sel = (mask >> kb) & 1u;
    const bf16_t* Ap = sel ? qb16 : lq16;
    const bf16_t* Bp = sel ? kb16 : lk16;
    float sc = sel ? SCALE : 1.0f;
    // ---- score/L tile [128x64] via WMMA (K=D=128)
#pragma unroll
    for (int nt = 0; nt < 4; ++nt) {
      v8f acc = {};
#pragma unroll
      for (int ks = 0; ks < 4; ++ks) {
        v16bf a = frag_a(Ap + ((size_t)hh * SEQ + qrow0 + m0) * HD + ks * 32, HD, lane);
        v16bf b = frag_bt(Bp + ((size_t)hh * SEQ + kb * BLKK + nt * 16) * HD + ks * 32,
                          HD, lane);
        acc = wmma_bf16(a, b, acc);
      }
      int mr = m0 + ((lane & 16) ? 8 : 0);
      int n  = lane & 15;
#pragma unroll
      for (int r = 0; r < 8; ++r) s_tile[mr + r][nt * 16 + n] = acc[r] * sc;
    }
    __syncthreads();
    // ---- per-row statistics pass; emits bf16 prob/L tile
    if (t < BLKQ) {
      if (sel) {
        float mx = NEGV;
        for (int c = 0; c < BLKK; ++c) mx = fmaxf(mx, s_tile[t][c]);
        float mold = s_m[t];
        float mnew = fmaxf(mold, mx);
        float resc = __expf(mold - mnew);
        float sum = 0.f;
        for (int c = 0; c < BLKK; ++c) {
          float p = __expf(s_tile[t][c] - mnew);
          p16[t][c] = (bf16_t)p;
          sum += p;
        }
        s_ds[t] = s_ds[t] * resc + sum;
        s_m[t]  = mnew;
        s_rs[t] = resc;
      } else {
        float sum = 0.f;
        for (int c = 0; c < BLKK; ++c) {
          float p = s_tile[t][c];
          p16[t][c] = (bf16_t)p;
          sum += p;
        }
        s_dl[t] += sum;
      }
    }
    __syncthreads();
    // ---- P·V (selected, rescaled) or L·V (unselected) accumulate
    int rbase = m0 + ((lane & 16) ? 8 : 0);
    if (sel) {
      float rs[8];
#pragma unroll
      for (int r = 0; r < 8; ++r) rs[r] = s_rs[rbase + r];
#pragma unroll
      for (int nt = 0; nt < 8; ++nt) {
        v8f acc = acc_s[nt];
#pragma unroll
        for (int r = 0; r < 8; ++r) acc[r] *= rs[r];
#pragma unroll
        for (int ks = 0; ks < 2; ++ks) {
          v16bf a = frag_a(&p16[m0][0] + ks * 32, PTS, lane);
          v16bf b = frag_bt(vt + ((size_t)hh * HD + nt * 16) * SEQ + kb * BLKK + ks * 32,
                            SEQ, lane);
          acc = wmma_bf16(a, b, acc);
        }
        acc_s[nt] = acc;
      }
    } else {
#pragma unroll
      for (int nt = 0; nt < 8; ++nt) {
        v8f acc = acc_l[nt];
#pragma unroll
        for (int ks = 0; ks < 2; ++ks) {
          v16bf a = frag_a(&p16[m0][0] + ks * 32, PTS, lane);
          v16bf b = frag_bt(vt + ((size_t)hh * HD + nt * 16) * SEQ + kb * BLKK + ks * 32,
                            SEQ, lane);
          acc = wmma_bf16(a, b, acc);
        }
        acc_l[nt] = acc;
      }
    }
    __syncthreads();
  }
  // ---- epilogue
  int rbase = m0 + ((lane & 16) ? 8 : 0);
  int n = lane & 15;
#pragma unroll
  for (int r = 0; r < 8; ++r) {
    int row = rbase + r;
    float w   = __expf(-s_m[row]);
    float den = s_ds[row] + w * s_dl[row];
#pragma unroll
    for (int nt = 0; nt < 8; ++nt) {
      float val = (acc_s[nt][r] + w * acc_l[nt][r]) / den;
      obf[(size_t)(qrow0 + row) * DIMSZ + hh * HD + nt * 16 + n] = (bf16_t)val;
    }
  }
}

// ---------------- host side ----------------
extern "C" void kernel_launch(void* const* d_in, const int* in_sizes, int n_in,
                              void* d_out, int out_size, void* d_ws, size_t ws_size,
                              hipStream_t stream) {
  const float* hidden = (const float*)d_in[0];
  const float* fcos   = (const float*)d_in[1];
  const float* fsin   = (const float*)d_in[2];
  const float* Wq = (const float*)d_in[3];
  const float* bq = (const float*)d_in[4];
  const float* Wk = (const float*)d_in[5];
  const float* bk = (const float*)d_in[6];
  const float* Wv = (const float*)d_in[7];
  const float* bv = (const float*)d_in[8];
  const float* Wo = (const float*)d_in[9];
  const float* bo = (const float*)d_in[10];
  const float* gq = (const float*)d_in[11];
  const float* gk = (const float*)d_in[12];
  float* out = (float*)d_out;

  char* wsp = (char*)d_ws;
  size_t off = 0;
  auto alloc = [&](size_t bytes) -> void* {
    void* p = wsp + off;
    off = (off + bytes + 255) & ~(size_t)255;
    return p;
  };
  bf16_t* Xbf   = (bf16_t*)alloc((size_t)SEQ * DIMSZ * 2);
  bf16_t* Wqt   = (bf16_t*)alloc((size_t)DIMSZ * DIMSZ * 2);  // transposed [N][K]
  bf16_t* Wkt   = (bf16_t*)alloc((size_t)DIMSZ * DIMSZ * 2);
  bf16_t* Wvt   = (bf16_t*)alloc((size_t)DIMSZ * DIMSZ * 2);
  bf16_t* Wot   = (bf16_t*)alloc((size_t)DIMSZ * DIMSZ * 2);
  float*  qproj = (float*)alloc((size_t)SEQ * DIMSZ * 4);
  float*  kproj = (float*)alloc((size_t)SEQ * DIMSZ * 4);
  float*  vproj = (float*)alloc((size_t)SEQ * DIMSZ * 4);
  float*  qf    = (float*)alloc((size_t)NH * SEQ * HD * 4);
  float*  kf    = (float*)alloc((size_t)NH * SEQ * HD * 4);
  bf16_t* q16   = (bf16_t*)alloc((size_t)NH * SEQ * HD * 2);
  bf16_t* k16   = (bf16_t*)alloc((size_t)NH * SEQ * HD * 2);
  bf16_t* vt16  = (bf16_t*)alloc((size_t)NH * SEQ * HD * 2);  // [H][D][S]
  bf16_t* lq16  = (bf16_t*)alloc((size_t)NH * SEQ * HD * 2);
  bf16_t* lk16  = (bf16_t*)alloc((size_t)NH * SEQ * HD * 2);
  float*  qpool = (float*)alloc((size_t)NH * NQB * HD * 4);
  float*  kpool = (float*)alloc((size_t)NH * NKB * HD * 4);
  unsigned* maskb = (unsigned*)alloc((size_t)NH * NQB * 4);
  bf16_t* obf   = (bf16_t*)alloc((size_t)SEQ * DIMSZ * 2);

  cvt_bf16_kernel<<<1024, 256, 0, stream>>>(hidden, Xbf, SEQ * DIMSZ);
  dim3 tgrid(DIMSZ / 32, DIMSZ / 32);  // (48,48)
  cvt_transpose_kernel<<<tgrid, 256, 0, stream>>>(Wq, Wqt, DIMSZ, DIMSZ);
  cvt_transpose_kernel<<<tgrid, 256, 0, stream>>>(Wk, Wkt, DIMSZ, DIMSZ);
  cvt_transpose_kernel<<<tgrid, 256, 0, stream>>>(Wv, Wvt, DIMSZ, DIMSZ);
  cvt_transpose_kernel<<<tgrid, 256, 0, stream>>>(Wo, Wot, DIMSZ, DIMSZ);

  dim3 ggrid(DIMSZ / 64, SEQ / 128);  // (24, 16)
  gemm_bias_kernel<<<ggrid, 256, 0, stream>>>(Xbf, Wqt, bq, qproj, SEQ, DIMSZ, DIMSZ);
  gemm_bias_kernel<<<ggrid, 256, 0, stream>>>(Xbf, Wkt, bk, kproj, SEQ, DIMSZ, DIMSZ);
  gemm_bias_kernel<<<ggrid, 256, 0, stream>>>(Xbf, Wvt, bv, vproj, SEQ, DIMSZ, DIMSZ);

  qk_post_kernel<<<SEQ, 256, 0, stream>>>(qproj, gq, fcos, fsin, qf, q16);
  qk_post_kernel<<<SEQ, 256, 0, stream>>>(kproj, gk, fcos, fsin, kf, k16);
  v_post_kernel<<<SEQ, 256, 0, stream>>>(vproj, vt16);

  rowsm_kernel<<<NH * SEQ, 128, 0, stream>>>(qf, lq16);
  rowsm_kernel<<<NH * SEQ, 128, 0, stream>>>(kf, lk16);

  pool_kernel<<<NH * NQB, 128, 0, stream>>>(qf, qpool, BLKQ, NQB);
  pool_kernel<<<NH * NKB, 128, 0, stream>>>(kf, kpool, BLKK, NKB);
  topk_kernel<<<NH, 32, 0, stream>>>(qpool, kpool, maskb);

  sla_kernel<<<dim3(NH, NQB), 256, 0, stream>>>(q16, k16, vt16, lq16, lk16, maskb, obf);

  gemm_bias_kernel<<<ggrid, 256, 0, stream>>>(obf, Wot, bo, out, SEQ, DIMSZ, DIMSZ);
}
